// Model_64682207477811
// MI455X (gfx1250) — matrix operand order repaired and checked by reading, not compiled, not run on hardware
//
#include <hip/hip_runtime.h>
#include <math.h>

// ============================================================================
// CDNA5 (gfx1250). All matmuls -> v_wmma_f32_16x16x32_f16 (f32 accum).
// Every GEMM operand (weights + activations + attention Q/K/V/probs) lives in
// fragment-order f16; all tile staging is global_load_async_to_lds_b128
// (ASYNCcnt), double-buffered so copies overlap WMMA, one barrier per k-step.
// WMMA operands load from LDS as 2x ds_load_b128 per lane.
// f32 kept for BN stats, residual chain, softmax.
// Wave32: 256-thread blocks = 8 waves. Workspace: ~175 MB in d_ws.
// ============================================================================

typedef __attribute__((ext_vector_type(16))) _Float16 v16h;
typedef __attribute__((ext_vector_type(8)))  _Float16 v8h;
typedef __attribute__((ext_vector_type(8)))  float    v8f;

union HF16 { v16h v; v8h p[2]; _Float16 e[16]; };

__device__ __forceinline__ v8f wmma16(v16h a, v16h b, v8f c) {
  return __builtin_amdgcn_wmma_f32_16x16x32_f16(
      false, a, false, b, (short)0, c, false, false);
}

// Fragment region = 512 halves (1 KB), lane's 16 halves contiguous.
__device__ __forceinline__ v16h frag_ld(const _Float16* __restrict__ base, int lane) {
  HF16 u;
  const v8h* p = (const v8h*)base + lane * 2;
  u.p[0] = p[0];
  u.p[1] = p[1];
  return u.v;
}

// A-matrix 16x32 (MxK) fragment position for element (m,k).
__device__ __forceinline__ int a_frag_off(int m, int k) {
  return ((m & 15) + (((k >> 3) & 1) << 4)) * 16 + ((k & 7) + ((k >> 4) << 3));
}
// B-matrix 32x16 (KxN) fragment position for element (k,n).
__device__ __forceinline__ int b_frag_off(int k, int n) {
  return ((n & 15) + (((k >> 3) & 1) << 4)) * 16 + ((k & 7) + ((k >> 4) << 3));
}
// Swizzled conv-input layout: [b][kchunk][nchunk(p/16)][512 halves].
__device__ __forceinline__ long sw_off(long bKC16384, int ch, int p) {
  return bKC16384 + (((ch >> 5) * 32 + (p >> 4)) << 9) + b_frag_off(ch & 31, p & 15);
}

__device__ __forceinline__ void st2h(_Float16* dst, float lo, float hi) {
  union { _Float16 h[2]; unsigned u; } t;
  t.h[0] = (_Float16)lo;
  t.h[1] = (_Float16)hi;
  *(unsigned*)dst = t.u;
}

// Async global->LDS 16-byte copy (per lane), tracked by ASYNCcnt.
__device__ __forceinline__ void async_copy_b128(unsigned lds_byte_off,
                                                const _Float16* g) {
  asm volatile("global_load_async_to_lds_b128 %0, %1, off"
               :: "v"(lds_byte_off), "v"(g) : "memory");
}
__device__ __forceinline__ void wait_async0() {
  asm volatile("s_wait_asynccnt 0x0" ::: "memory");
}
__device__ __forceinline__ unsigned lds_addr(const void* p) {
  return (unsigned)(unsigned long long)p;
}

// ============================================================================
// Weight pre-swizzle: W (MxK f32) -> WH fragment order [kchunk][mchunk][512],
// zero-padded to Mpad = MC16*16 (multiple of 64), Kpad (multiple of 32).
// ============================================================================
__global__ __launch_bounds__(256) void weight_swizzle(
    const float* __restrict__ W, _Float16* __restrict__ WH,
    int M, int K, int MC16, int Kpad) {
  long total = (long)MC16 * 16 * Kpad;
  for (long i = (long)blockIdx.x * 256 + threadIdx.x; i < total;
       i += (long)gridDim.x * 256) {
    int k = (int)(i % Kpad);
    int m = (int)(i / Kpad);
    float v = (m < M && k < K) ? W[(long)m * K + k] : 0.0f;
    WH[(((long)(k >> 5) * MC16 + (m >> 4)) << 9) + a_frag_off(m & 15, k & 31)] =
        (_Float16)v;
  }
}

__global__ __launch_bounds__(256) void fill_zero_h(_Float16* __restrict__ p, long n) {
  for (long i = (long)blockIdx.x * 256 + threadIdx.x; i < n;
       i += (long)gridDim.x * 256)
    p[i] = (_Float16)0.0f;
}

// ============================================================================
// Batched GEMM: out[b] (MxN) = WH (swizzled) * Xs[b] (swizzled) + bias.
// Tile 64x128; double-buffered async staging (A: 4KB, B: 8KB per k-step).
// outMode 0: plain f32 Y (+res, + optional swizzled conv-input dual write)
// outMode 1/2/3: write Q / K / V attention fragment layouts (f16 only).
// ============================================================================
__global__ __launch_bounds__(256) void conv_gemm(
    const _Float16* __restrict__ WH, const float* __restrict__ bias,
    const _Float16* __restrict__ Xs, int KC, int MC16,
    const float* __restrict__ res, float* __restrict__ Y,
    _Float16* __restrict__ Ysw, int KCout,
    int outMode, int H, int CC16, int M, int N) {
  __shared__ __align__(16) _Float16 Af[2][4 * 512];   // 2 x 4 KB
  __shared__ __align__(16) _Float16 Bf[2][8 * 512];   // 2 x 8 KB
  int tid = threadIdx.x, lane = tid & 31, w = tid >> 5;
  int nBase = blockIdx.x * 128, mBase = blockIdx.y * 64;
  long bOffH = (long)blockIdx.z * KC * 16384;
  long yOff  = (long)blockIdx.z * M * N;
  long oOffH = (long)blockIdx.z * KCout * 16384;
  int mFrag = w >> 1;
  int nGrp  = (w & 1) * 4;
  int Kpad  = KC * 32;
  v8f acc[4] = {{}, {}, {}, {}};

  auto issueAB = [&](int k0, int buf) {
    const _Float16* ga =
        WH + (((long)(k0 >> 5) * MC16 + (mBase >> 4)) << 9) + tid * 8;
    async_copy_b128(lds_addr(&Af[buf][0]) + (unsigned)tid * 16u, ga);
    const _Float16* gb =
        Xs + bOffH + (((long)(k0 >> 5) * 32 + (nBase >> 4)) << 9) + tid * 8;
    async_copy_b128(lds_addr(&Bf[buf][0]) + (unsigned)tid * 16u, gb);
    async_copy_b128(lds_addr(&Bf[buf][0]) + (unsigned)tid * 16u + 4096u, gb + 2048);
  };

  issueAB(0, 0);
  for (int k0 = 0; k0 < Kpad; k0 += 32) {
    int cur = (k0 >> 5) & 1;
    wait_async0();
    __syncthreads();
    if (k0 + 32 < Kpad) issueAB(k0 + 32, cur ^ 1);
    v16h a = frag_ld(&Af[cur][0] + mFrag * 512, lane);
#pragma unroll
    for (int s = 0; s < 4; ++s) {
      v16h b = frag_ld(&Bf[cur][0] + (nGrp + s) * 512, lane);
      acc[s] = wmma16(a, b, acc[s]);
    }
  }

  int mLoc = (lane >> 4) * 8, nLoc = lane & 15;
#pragma unroll
  for (int s = 0; s < 4; ++s) {
    int gn = nBase + (nGrp + s) * 16 + nLoc;
#pragma unroll
    for (int vi = 0; vi < 8; ++vi) {
      int gm = mBase + mFrag * 16 + mLoc + vi;
      if (gm >= M) continue;
      float vv = acc[s][vi] + (bias ? bias[gm] : 0.0f);
      if (outMode == 0) {
        long idx = yOff + (long)gm * N + gn;
        if (res) vv += res[idx];
        Y[idx] = vv;
        if (Ysw) Ysw[sw_off(oOffH, gm, gn)] = (_Float16)vv;
      } else {
        int hh = gm % H, c = gm / H;
        long bh = (long)blockIdx.z * H + hh;
        long off;
        if (outMode == 1)        // Q: A-frag over (m=i, k=c)
          off = ((bh * 32 + (gn >> 4)) * KC + (c >> 5)) * 512 +
                a_frag_off(gn & 15, c & 31);
        else if (outMode == 2)   // K: B-frag over (k=c, n=j)
          off = ((bh * KC + (c >> 5)) * 32 + (gn >> 4)) * 512 +
                b_frag_off(c & 31, gn & 15);
        else                     // V: A-frag over (m=c, k=j)
          off = ((bh * 16 + (gn >> 5)) * CC16 + (c >> 4)) * 512 +
                a_frag_off(c & 15, gn & 31);
        Ysw[off] = (_Float16)vv;
      }
    }
  }
}

// ============================================================================
// Similarity: per b, Out[i,j] = sigmoid(sum_c X[c,i]*X[c,j]); C=128, P=512.
// Reads plain f32 x. Tile 64(i) x 128(j).
// ============================================================================
__global__ __launch_bounds__(256) void sim_gemm(const float* __restrict__ Xin,
                                                float* __restrict__ Out) {
  __shared__ __align__(16) _Float16 Af[4 * 512];
  __shared__ __align__(16) _Float16 Bf[8 * 512];
  int tid = threadIdx.x, lane = tid & 31, w = tid >> 5;
  int jBase = blockIdx.x * 128, iBase = blockIdx.y * 64;
  const float* Xb = Xin + (long)blockIdx.z * 128 * 512;
  int mFrag = w >> 1;
  int nGrp  = (w & 1) * 4;
  v8f acc[4] = {{}, {}, {}, {}};
  for (int c0 = 0; c0 < 128; c0 += 32) {
    if (c0 + 32 < 128)
      __builtin_prefetch(&Xb[(long)(c0 + 32) * 512 + jBase + (tid & 127)], 0, 1);
    for (int t = tid; t < 1024; t += 256) {
      int kc = (t >> 6) * 2, mi = t & 63;
      const float* src = Xb + (long)(c0 + kc) * 512 + iBase + mi;
      st2h(&Af[(mi >> 4) * 512 + a_frag_off(mi & 15, kc)], src[0], src[512]);
    }
    for (int t = tid; t < 2048; t += 256) {
      int kc = (t >> 7) * 2, nj = t & 127;
      const float* src = Xb + (long)(c0 + kc) * 512 + jBase + nj;
      st2h(&Bf[(nj >> 4) * 512 + b_frag_off(kc, nj & 15)], src[0], src[512]);
    }
    __syncthreads();
    v16h a = frag_ld(Af + mFrag * 512, lane);
#pragma unroll
    for (int s = 0; s < 4; ++s) {
      v16h b = frag_ld(Bf + (nGrp + s) * 512, lane);
      acc[s] = wmma16(a, b, acc[s]);
    }
    __syncthreads();
  }
  int mLoc = (lane >> 4) * 8, nLoc = lane & 15;
  long oOff = (long)blockIdx.z * 512 * 512;
#pragma unroll
  for (int s = 0; s < 4; ++s) {
    int gj = jBase + (nGrp + s) * 16 + nLoc;
#pragma unroll
    for (int vi = 0; vi < 8; ++vi) {
      int gi = iBase + mFrag * 16 + mLoc + vi;
      Out[oOff + (long)gi * 512 + gj] = 1.0f / (1.0f + __expf(-acc[s][vi]));
    }
  }
}

// ============================================================================
// Fused attention: per (b, head, 16-row i-tile). Q/K/V arrive pre-swizzled
// f16; all staging is double-buffered async b128 copies.
// Output h written in swizzled conv-input format (ch = c*H + hh).
// ============================================================================
__global__ __launch_bounds__(256) void attn_kernel(
    const _Float16* __restrict__ qa, const _Float16* __restrict__ kb,
    const _Float16* __restrict__ va, _Float16* __restrict__ hsw,
    int C, int H, int KC, int CC16, float scale) {
  __shared__ __align__(16) float    Sf32[16 * 512];   // 32 KB (-> 2x V bufs)
  __shared__ __align__(16) _Float16 RegionB[2 * 4608]; // ph1 2 bufs; ph2/3 probs
  __shared__ float red[256];

  int tid = threadIdx.x, lane = tid & 31, w = tid >> 5;
  int i0 = blockIdx.x * 16;
  int hh = blockIdx.y;
  int b  = blockIdx.z;
  long bh = (long)b * H + hh;
  int Kpad = KC * 32;

  // ---- Phase 1: scores (flat pipelined (j0, c0) sequence) ----
  auto issueQK = [&](int j0, int c0, int buf) {
    _Float16* Aq = RegionB + buf * 4608;
    if (tid < 64)
      async_copy_b128(lds_addr(Aq) + (unsigned)tid * 16u,
                      qa + ((bh * 32 + (i0 >> 4)) * KC + (c0 >> 5)) * 512 + tid * 8);
    const _Float16* gb =
        kb + ((bh * KC + (c0 >> 5)) * 32 + (j0 >> 4)) * 512 + tid * 8;
    async_copy_b128(lds_addr(Aq + 512) + (unsigned)tid * 16u, gb);
    async_copy_b128(lds_addr(Aq + 512) + (unsigned)tid * 16u + 4096u, gb + 2048);
  };
  {
    int nsteps = 4 * KC;
    issueQK(0, 0, 0);
    v8f acc = {};
    int jc = 0, cc = 0, buf = 0;
    for (int s = 0; s < nsteps; ++s) {
      wait_async0();
      __syncthreads();
      int jn = jc, cn = cc + 32;
      if (cn == Kpad) { cn = 0; jn += 128; }
      if (s + 1 < nsteps) issueQK(jn, cn, buf ^ 1);
      _Float16* Aq = RegionB + buf * 4608;
      v16h af = frag_ld(Aq, lane);
      v16h bf = frag_ld(Aq + 512 + w * 512, lane);
      acc = wmma16(af, bf, acc);
      if (cn == 0) {  // finished a 128-wide j block
        int mLoc = (lane >> 4) * 8, nLoc = lane & 15;
#pragma unroll
        for (int vi = 0; vi < 8; ++vi)
          Sf32[(mLoc + vi) * 512 + jc + w * 16 + nLoc] = acc[vi] * scale;
        v8f z = {};
        acc = z;
      }
      jc = jn; cc = cn; buf ^= 1;
    }
  }
  __syncthreads();

  // ---- Phase 2: softmax over 512 (16 rows x 16 threads) ----
  int r = tid >> 4, l16 = tid & 15;
  float* Srow = Sf32 + r * 512;
  float mx = -3.0e38f;
  for (int j = l16; j < 512; j += 16) mx = fmaxf(mx, Srow[j]);
  red[tid] = mx;
  __syncthreads();
#pragma unroll
  for (int t = 0; t < 16; ++t) mx = fmaxf(mx, red[r * 16 + t]);
  __syncthreads();
  float sum = 0.0f;
  for (int j = l16; j < 512; j += 16) {
    float e = __expf(Srow[j] - mx);
    Srow[j] = e;
    sum += e;
  }
  red[tid] = sum;
  __syncthreads();
  sum = 0.0f;
#pragma unroll
  for (int t = 0; t < 16; ++t) sum += red[r * 16 + t];
  float inv = 1.0f / sum;
  for (int j = l16; j < 512; j += 16) {
    int jj = j & 31;
    RegionB[(j >> 5) * 512 + b_frag_off(jj, r)] = (_Float16)(Srow[j] * inv);
  }
  __syncthreads();

  // ---- Phase 3: H = V * S^T (M=C pad 128, N=16, K=512; pipelined k=64) ----
  int KCh = (C * H) >> 5;
  long hOffH = (long)b * KCh * 16384;
  auto issueV = [&](int j0, int m0, int buf) {
    _Float16* Vb = (_Float16*)Sf32 + buf * 8192;  // 16 KB per buffer
#pragma unroll
    for (int s = 0; s < 2; ++s) {
      const _Float16* gv =
          va + ((bh * 16 + (j0 >> 5) + s) * CC16 + (m0 >> 4)) * 512 + tid * 8;
      async_copy_b128(lds_addr(Vb) + (unsigned)(s * 8192 + tid * 16), gv);
      async_copy_b128(lds_addr(Vb) + (unsigned)(s * 8192 + tid * 16 + 4096),
                      gv + 2048);
    }
  };
  for (int m0 = 0; m0 < C; m0 += 128) {
    issueV(0, m0, 0);
    v8f acc = {};
    for (int j0 = 0; j0 < 512; j0 += 64) {
      int cur = (j0 >> 6) & 1;
      wait_async0();
      __syncthreads();
      if (j0 + 64 < 512) issueV(j0 + 64, m0, cur ^ 1);
      _Float16* Vb = (_Float16*)Sf32 + cur * 8192;
#pragma unroll
      for (int s = 0; s < 2; ++s) {
        v16h af = frag_ld(Vb + (s * 8 + w) * 512, lane);
        v16h bf = frag_ld(RegionB + ((j0 >> 5) + s) * 512, lane);
        acc = wmma16(af, bf, acc);
      }
    }
    int mLoc = (lane >> 4) * 8, nLoc = lane & 15;
#pragma unroll
    for (int vi = 0; vi < 8; ++vi) {
      int c = m0 + w * 16 + mLoc + vi;
      if (c < C) {
        int ch = c * H + hh;
        hsw[sw_off(hOffH, ch, i0 + nLoc)] = (_Float16)acc[vi];
      }
    }
  }
}

// ============================================================================
// BatchNorm stats: one block per channel; mean/rstd over (b=32, p=512).
// ============================================================================
__global__ __launch_bounds__(256) void bn_stats(const float* __restrict__ X,
                                                int C, float* __restrict__ mean,
                                                float* __restrict__ rstd) {
  __shared__ float s1[256], s2[256];
  int c = blockIdx.x;
  float a = 0.0f, qq = 0.0f;
  for (int t = threadIdx.x; t < 32 * 512; t += 256) {
    int bb = t >> 9, p = t & 511;
    float x = X[(((long)bb * C) + c) * 512 + p];
    a += x; qq += x * x;
  }
  s1[threadIdx.x] = a; s2[threadIdx.x] = qq;
  __syncthreads();
  for (int off = 128; off > 0; off >>= 1) {
    if (threadIdx.x < off) {
      s1[threadIdx.x] += s1[threadIdx.x + off];
      s2[threadIdx.x] += s2[threadIdx.x + off];
    }
    __syncthreads();
  }
  if (threadIdx.x == 0) {
    float m = s1[0] / 16384.0f;
    float var = s2[0] / 16384.0f - m * m;
    mean[c] = m;
    rstd[c] = rsqrtf(var + 1e-5f);
  }
}

// BN apply (+silu, +res). Writes plain f32 (optional) and swizzled f16
// (channel-padded to Cpad = KCx*32; pad channels written as 0).
__global__ __launch_bounds__(256) void bn_apply(
    const float* __restrict__ Yc, const float* __restrict__ mean,
    const float* __restrict__ rstd, const float* __restrict__ g,
    const float* __restrict__ beta, const float* __restrict__ res,
    float* __restrict__ out, _Float16* __restrict__ outsw,
    int C, int KCx, int do_silu) {
  int Cpad = KCx * 32;
  int idx = blockIdx.x * 256 + threadIdx.x;
  if (idx >= 32 * Cpad * 512) return;
  int p = idx & 511;
  int c = (idx >> 9) % Cpad;
  int b = (idx >> 9) / Cpad;
  long hOff = (long)b * KCx * 16384;
  if (c >= C) {
    outsw[sw_off(hOff, c, p)] = (_Float16)0.0f;
    return;
  }
  long pidx = (((long)b * C) + c) * 512 + p;
  float v = (Yc[pidx] - mean[c]) * rstd[c] * g[c] + beta[c];
  if (do_silu) v = v / (1.0f + __expf(-v));
  if (res) v += res[pidx];
  if (out) out[pidx] = v;
  outsw[sw_off(hOff, c, p)] = (_Float16)v;
}

// Build stage-0 input directly in swizzled f16 (C=4, Cpad=32).
__global__ __launch_bounds__(256) void prep_kernel(
    const float* __restrict__ in, const int* __restrict__ parts,
    const float* __restrict__ n1, _Float16* __restrict__ Xs) {
  int idx = blockIdx.x * 256 + threadIdx.x;
  if (idx >= 32 * 32 * 512) return;
  int p = idx & 511;
  int c = (idx >> 9) & 31;
  int b = idx >> 14;
  float v = 0.0f;
  if (c < 3) v = in[((long)b * 512 + p) * 3 + c];
  else if (c == 3) v = (float)parts[b] + n1[(long)b * 512 + p];
  Xs[sw_off((long)b * 16384, c, p)] = (_Float16)v;
}

// Build stage-3 input [32, 513->544, 512] directly in swizzled f16.
__global__ __launch_bounds__(256) void concat_kernel(
    const float* __restrict__ sim, const int* __restrict__ parts,
    const float* __restrict__ n2, _Float16* __restrict__ Xs) {
  int idx = blockIdx.x * 256 + threadIdx.x;
  if (idx >= 32 * 544 * 512) return;
  int p = idx & 511;
  int c = (idx >> 9) % 544;
  int b = (idx >> 9) / 544;
  float v = 0.0f;
  if (c < 512) v = sim[(((long)b * 512) + c) * 512 + p];
  else if (c == 512) v = (float)parts[b] + n2[(long)b * 512 + p];
  Xs[sw_off((long)b * 17 * 16384, c, p)] = (_Float16)v;
}

__global__ __launch_bounds__(256) void final_softmax(
    const float* __restrict__ X, float* __restrict__ out) {
  int idx = blockIdx.x * 256 + threadIdx.x;
  if (idx >= 32 * 512) return;
  int b = idx >> 9, p = idx & 511;
  float v[12];
  float mx = -3.0e38f;
#pragma unroll
  for (int c = 0; c < 12; ++c) {
    v[c] = X[(((long)b * 12) + c) * 512 + p];
    mx = fmaxf(mx, v[c]);
  }
  float s = 0.0f;
#pragma unroll
  for (int c = 0; c < 12; ++c) { v[c] = __expf(v[c] - mx); s += v[c]; }
  float inv = 1.0f / s;
#pragma unroll
  for (int c = 0; c < 12; ++c)
    out[(((long)b * 512) + p) * 12 + c] = v[c] * inv;
}

// ============================================================================
static inline int cdiv(int a, int b) { return (a + b - 1) / b; }

extern "C" void kernel_launch(void* const* d_in, const int* in_sizes, int n_in,
                              void* d_out, int out_size, void* d_ws, size_t ws_size,
                              hipStream_t stream) {
  (void)in_sizes; (void)n_in; (void)out_size; (void)ws_size;
  const float* inX   = (const float*)d_in[0];
  const int*   parts = (const int*)d_in[1];
  const float* no1   = (const float*)d_in[2];
  const float* no2   = (const float*)d_in[3];
  int pi = 4;  // params flattened after, jax-pytree order (sorted dict keys)
  auto F = [&](int i) { return (const float*)d_in[i]; };

  // Workspace layout. Requires ~175 MB.
  float* ws = (float*)d_ws;
  float* X    = ws;                          // plain f32 x
  float* Y    = X + 32L * 513 * 512;         // conv out (pre-BN)
  float* mean = Y + 32L * 256 * 512;
  float* rstd = mean + 512;
  _Float16* XH  = (_Float16*)(rstd + 512);   // swizzled x (<=17 kchunks)
  _Float16* XNH = XH + 32L * 17 * 16384;     // swizzled xn (<=8 kchunks)
  _Float16* HbH = XNH + 32L * 8 * 16384;     // swizzled attn h (<=12 kchunks)
  _Float16* QA  = HbH + 32L * 12 * 16384;    // Q A-frags, max 8.39M halves
  _Float16* KBs = QA + 8388608L;             // K B-frags
  _Float16* VA  = KBs + 8388608L;            // V A-frags, max 25.2M halves
  _Float16* WH  = VA + 25165824L;            // swizzled weights, <=160K halves

  float* simOut = (float*)d_out;             // [32,512,512]
  float* outSm  = simOut + 32L * 512 * 512;  // [32,512,12]

  prep_kernel<<<cdiv(32 * 32 * 512, 256), 256, 0, stream>>>(inX, parts, no1, XH);

  auto swizW = [&](const float* W, int M, int K) {
    int MC16 = cdiv(M, 64) * 4, Kpad = cdiv(K, 32) * 32;
    weight_swizzle<<<cdiv(MC16 * 16 * Kpad, 256), 256, 0, stream>>>(
        W, WH, M, K, MC16, Kpad);
    return MC16;
  };

  // mlp block: x = silu(bn(conv(x))) (+x if Cin==Cout)
  auto mlp = [&](int Cin, int Cout) {
    const float* W  = F(pi + 0);  // sorted keys: W, b, beta, g
    const float* bb = F(pi + 1);
    const float* be = F(pi + 2);
    const float* g  = F(pi + 3);
    pi += 4;
    int KCin = cdiv(Cin, 32), KCout = cdiv(Cout, 32);
    int MC16 = swizW(W, Cout, Cin);
    conv_gemm<<<dim3(4, cdiv(Cout, 64), 32), 256, 0, stream>>>(
        WH, bb, XH, KCin, MC16, nullptr, Y, nullptr, 0, 0, 1, 0, Cout, 512);
    bn_stats<<<Cout, 256, 0, stream>>>(Y, Cout, mean, rstd);
    bn_apply<<<cdiv(32 * KCout * 32 * 512, 256), 256, 0, stream>>>(
        Y, mean, rstd, g, be, (Cin == Cout) ? X : nullptr, X, XH, Cout, KCout, 1);
  };

  // attn block
  auto attn = [&](int C, int H) {
    // sorted keys: Wk, Wo, Wq, Wv, beta, bk, bo, bq, bv, g
    const float* Wk = F(pi + 0); const float* Wo = F(pi + 1);
    const float* Wq = F(pi + 2); const float* Wv = F(pi + 3);
    const float* be = F(pi + 4); const float* bk = F(pi + 5);
    const float* bo = F(pi + 6); const float* bq = F(pi + 7);
    const float* bv = F(pi + 8); const float* g  = F(pi + 9);
    pi += 10;
    int KC = cdiv(C, 32);
    int CC16 = (C > 128) ? 16 : 8;
    int CH = C * H;
    bn_stats<<<C, 256, 0, stream>>>(X, C, mean, rstd);
    bn_apply<<<cdiv(32 * KC * 32 * 512, 256), 256, 0, stream>>>(
        X, mean, rstd, g, be, nullptr, nullptr, XNH, C, KC, 0);  // xn (swizzled)
    if (C & 31) {  // zero Q/K pad channels (C=12 case)
      long n = 32L * H * 32 * KC * 512;
      fill_zero_h<<<2048, 256, 0, stream>>>(QA, n);
      fill_zero_h<<<2048, 256, 0, stream>>>(KBs, n);
    }
    int MC16 = swizW(Wq, CH, C);
    conv_gemm<<<dim3(4, cdiv(CH, 64), 32), 256, 0, stream>>>(
        WH, bq, XNH, KC, MC16, nullptr, nullptr, QA, 0, 1, H, CC16, CH, 512);
    swizW(Wk, CH, C);
    conv_gemm<<<dim3(4, cdiv(CH, 64), 32), 256, 0, stream>>>(
        WH, bk, XNH, KC, MC16, nullptr, nullptr, KBs, 0, 2, H, CC16, CH, 512);
    swizW(Wv, CH, C);
    conv_gemm<<<dim3(4, cdiv(CH, 64), 32), 256, 0, stream>>>(
        WH, bv, XNH, KC, MC16, nullptr, nullptr, VA, 0, 3, H, CC16, CH, 512);
    float scale = 1.0f / sqrtf((float)C);
    attn_kernel<<<dim3(32, H, 32), 256, 0, stream>>>(
        QA, KBs, VA, HbH, C, H, KC, CC16, scale);
    // out-proj: x = Wo*h + bo + x  (plain f32 + swizzled conv-input dual write)
    int MC16o = swizW(Wo, C, CH);
    conv_gemm<<<dim3(4, cdiv(C, 64), 32), 256, 0, stream>>>(
        WH, bo, HbH, CH / 32, MC16o, X, X, XH, KC, 0, 1, 0, C, 512);
  };

  // Stage 0
  mlp(4, 32);   attn(32, 12);  mlp(32, 32);  attn(32, 12);
  // Stage 1
  mlp(32, 64);  attn(64, 8);   mlp(64, 64);  attn(64, 8);
  mlp(64, 64);  attn(64, 8);
  // Stage 2
  mlp(64, 128); attn(128, 4);  mlp(128, 128); attn(128, 4);
  mlp(128, 128); attn(128, 4);
  // Similarity -> first output, then stage-3 input (swizzled, 544 channels)
  sim_gemm<<<dim3(4, 8, 32), 256, 0, stream>>>(X, simOut);
  concat_kernel<<<cdiv(32 * 544 * 512, 256), 256, 0, stream>>>(simOut, parts, no2, XH);
  // Stage 3
  mlp(513, 256); attn(256, 1); mlp(256, 128); attn(128, 4);
  mlp(128, 12);  attn(12, 8);
  // Final channel softmax -> second output
  final_softmax<<<cdiv(32 * 512, 256), 256, 0, stream>>>(X, outSm);
}
